// SSMKernelDiag_49254684950575
// MI455X (gfx1250) — matrix operand order repaired and checked by reading, not compiled, not run on hardware
//
#include <hip/hip_runtime.h>
#include <hip/hip_bf16.h>
#include <math.h>

// CDNA5 (gfx1250) wave32 WMMA types
typedef __attribute__((ext_vector_type(16))) _Float16 v16h;
typedef __attribute__((ext_vector_type(8)))  float    v8f;

union V16H { v16h v; _Float16 e[16]; };
union V8F  { v8f  v; float     e[8];  };

// S4D diagonal SSM kernel via Vandermonde factorization:
//   l = 16p + q,  exp(dtA*l) = exp(dtA*16p) * exp(dtA*q)
//   K[h, 256t + 16m + j] = sum_{k<2*Nh} A'[m][k] * B'[k][j]   (per head, per tile)
// computed with v_wmma_f32_16x16x32_f16 (f32 accumulate), one WMMA per chunk of
// 16 complex state dims (K=32 real).
//
// Fast-math transcendentals (v_exp_f32 / v_sin_f32 / v_cos_f32): their ~1e-3
// absolute phase error at the largest arguments matches the f16 operand
// quantization error of the WMMA inputs, so precise libm (Payne-Hanek) buys
// nothing and costs ~2.5k VALU ops of branchy glue.
template<int NH_CT>   // NH_CT > 0: compile-time Nh (full unroll); 0: runtime
__global__ __launch_bounds__(256) void s4d_wmma_kernel(
    const float* __restrict__ inv_dt,      // (H,1)
    const float* __restrict__ A_log_real,  // (H,Nh)
    const float* __restrict__ A_imag,      // (H,Nh)
    const float* __restrict__ B_re,        // (H,Nh)
    const float* __restrict__ B_im,        // (H,Nh)
    const float* __restrict__ C_re,        // (C,H,Nh)
    const float* __restrict__ C_im,        // (C,H,Nh)
    float* __restrict__ out,               // (C,H,L)
    int H, int Nh_rt, int Cch, int L)
{
    const int Nh   = (NH_CT > 0) ? NH_CT : Nh_rt;
    const int h    = blockIdx.x % H;
    const int cch  = blockIdx.x / H;
    const int lane = threadIdx.x & 31;
    const int wave = threadIdx.x >> 5;
    const int half = lane >> 4;    // which 16-lane half of the wave
    const int m16  = lane & 15;    // A row / B col / D col index
    const int nwaves = blockDim.x >> 5;

    const int hNh = h * Nh;
    const int cNh = (cch * H + h) * Nh;
    const float dt = __expf(inv_dt[h]);

    const int ntiles = L >> 8;     // each tile covers 256 consecutive lags
    float* outp = out + (size_t)(cch * H + h) * (size_t)L;

    for (int t = wave; t < ntiles; t += nwaves) {
        V8F acc;
        #pragma unroll
        for (int r = 0; r < 8; ++r) acc.e[r] = 0.0f;

        #pragma unroll
        for (int nOff = 0; nOff < Nh; nOff += 16) {
            // ---- A fragment (16x32 f16): row m = p - 16t, element pair g.
            // element e <-> K = 2*(e&8) + 8*half + (e&7); pair g covers
            // K = {2n_l, 2n_l+1} with n_l = (g<4?0:8) + 4*half + (g&3).
            V16H a;
            const float p16 = 16.0f * (float)(16 * t + m16);   // 16*p
            #pragma unroll
            for (int g = 0; g < 8; ++g) {
                const int n = nOff + ((g < 4) ? 0 : 8) + 4 * half + (g & 3);
                const float ReA = -__expf(A_log_real[hNh + n]);
                const float ImA = A_imag[hNh + n];
                const float ar = dt * ReA, ai = dt * ImA;      // dtA
                // dB = B * (exp(dtA)-1)/A ; CB = C * dB
                const float ew  = __expf(ar);
                const float sw  = __sinf(ai), cw = __cosf(ai);
                const float wre = ew * cw - 1.0f;
                const float wim = ew * sw;
                const float inv = 1.0f / (ReA * ReA + ImA * ImA);
                const float tre = (wre * ReA + wim * ImA) * inv;
                const float tim = (wim * ReA - wre * ImA) * inv;
                const float bre = B_re[hNh + n], bim = B_im[hNh + n];
                const float dre  = bre * tre - bim * tim;
                const float dim_ = bre * tim + bim * tre;
                const float cre = C_re[cNh + n], cim = C_im[cNh + n];
                const float CBre = cre * dre - cim * dim_;
                const float CBim = cre * dim_ + cim * dre;
                // G = CB * exp(dtA * 16p)
                const float eg = __expf(ar * p16);
                const float sg = __sinf(ai * p16), cg = __cosf(ai * p16);
                const float Gre = eg * (CBre * cg - CBim * sg);
                const float Gim = eg * (CBre * sg + CBim * cg);
                a.e[2 * g]     = (_Float16)( 2.0f * Gre);
                a.e[2 * g + 1] = (_Float16)(-2.0f * Gim);
            }

            // ---- B fragment (32x16 f16): col q = lane&15.
            // element e <-> K = 2*(e&8) + 8*half + (e&7); pair g -> n_l = (g<4?0:8) + 4*half + (g&3).
            V16H b;
            const float fq = (float)m16;
            #pragma unroll
            for (int g = 0; g < 8; ++g) {
                const int n = nOff + ((g < 4) ? 0 : 8) + 4 * half + (g & 3);
                const float ar = dt * (-__expf(A_log_real[hNh + n]));
                const float ai = dt * A_imag[hNh + n];
                const float eu = __expf(ar * fq);
                const float su = __sinf(ai * fq), cu = __cosf(ai * fq);
                b.e[2 * g]     = (_Float16)(eu * cu);   // Re(exp(q*dtA))
                b.e[2 * g + 1] = (_Float16)(eu * su);   // Im(exp(q*dtA))
            }

            // D = A' x B' + D   (16x16x32, f32 accumulate)
            acc.v = __builtin_amdgcn_wmma_f32_16x16x32_f16(
                false, a.v, false, b.v, (short)0, acc.v, false, false);
        }

        // ---- store: D element r -> l = 256t + 16*(r + 8*half) + (lane&15)
        float* o = outp + 256 * t + 128 * half + m16;
        #pragma unroll
        for (int r = 0; r < 8; ++r) o[16 * r] = acc.e[r];
    }
}

extern "C" void kernel_launch(void* const* d_in, const int* in_sizes, int n_in,
                              void* d_out, int out_size, void* d_ws, size_t ws_size,
                              hipStream_t stream) {
    (void)n_in; (void)d_ws; (void)ws_size;
    const float* inv_dt = (const float*)d_in[0];
    const float* Alr    = (const float*)d_in[1];
    const float* Aim    = (const float*)d_in[2];
    const float* Bre    = (const float*)d_in[3];
    const float* Bim    = (const float*)d_in[4];
    const float* Cre    = (const float*)d_in[5];
    const float* Cim    = (const float*)d_in[6];
    float* out = (float*)d_out;

    // Derive shapes host-side (d_in[7] holds L on device; unreadable under capture).
    const int H   = in_sizes[0];              // inv_dt is (H,1)
    const int Nh  = in_sizes[1] / H;          // A_log_real is (H,Nh)
    const int Cch = in_sizes[5] / (H * Nh);   // C_re is (C,H,Nh)
    const int L   = out_size / (Cch * H);     // out is (C,H,L)

    dim3 grid(H * Cch), block(256);           // 8 wave32 per head; 1 wave = 1 tile
    if (Nh == 32) {
        s4d_wmma_kernel<32><<<grid, block, 0, stream>>>(
            inv_dt, Alr, Aim, Bre, Bim, Cre, Cim, out, H, Nh, Cch, L);
    } else {
        s4d_wmma_kernel<0><<<grid, block, 0, stream>>>(
            inv_dt, Alr, Aim, Bre, Bim, Cre, Cim, out, H, Nh, Cch, L);
    }
}
